// ESRNN_26809185862197
// MI455X (gfx1250) — compile-verified, matrix operand order checked by reading, not run on hardware
//
#include <hip/hip_runtime.h>
#include <hip/hip_bf16.h>
#include <stdint.h>

typedef __bf16 bf16_t;
typedef __attribute__((ext_vector_type(16))) __bf16 v16bf;
typedef __attribute__((ext_vector_type(8)))  float  v8f;

#define T1   511      // scan length (T-1)
#define BB   1024     // batch
#define HH   64       // hidden
#define NT   16       // N tiles (256/16)
#define LDT  72       // padded LDS row stride (elements, 144B = 16B aligned, conflict-free)
#define WPB  4        // waves per block in scan kernels (1 wave/SIMD -> 1024-VGPR budget)

// ---------------- fast transcendentals (v_exp_f32 / v_rcp_f32) ----------------
__device__ __forceinline__ float fast_sig(float x) {
    return __builtin_amdgcn_rcpf(1.f + __builtin_amdgcn_exp2f(-1.44269504f * x));
}
__device__ __forceinline__ float fast_tanh(float x) {
    float e = __builtin_amdgcn_exp2f(2.88539008f * x);   // exp(2x)
    return 1.f - 2.f * __builtin_amdgcn_rcpf(e + 1.f);
}

// Intra-fragment K mapping (16-bit A layout per CDNA5 ISA; B packed with the
// identical mapping so the WMMA dot-product is permutation-consistent).
__device__ __forceinline__ int kmap2(int i, int half) {   // K for element e=2i (pair base)
    return ((i < 4) ? 0 : 16) + half * 8 + ((i & 3) << 1);
}

union FragU { v16bf v; unsigned u[8]; };
union Pk16 { uint4 q; bf16_t b[8]; };

// =======================  embeddings -> projection tables  =======================
__global__ void catproj_kernel(const int* __restrict__ cid0, const int* __restrict__ cid1,
                               const int* __restrict__ cid2, const int* __restrict__ cid3,
                               const float* __restrict__ emb0, const float* __restrict__ emb1,
                               const float* __restrict__ emb2, const float* __restrict__ emb3,
                               const float* __restrict__ Wih0, const float* __restrict__ b_all,
                               float* __restrict__ catproj)
{
    __shared__ float feat[61];
    int b = blockIdx.x;
    int tid = threadIdx.x;
    if (tid < 61) {
        float v;
        if (tid < 50)      v = emb0[cid0[b] * 50 + tid];
        else if (tid < 54) v = emb1[cid1[b] * 4 + (tid - 50)];
        else if (tid < 59) v = emb2[cid2[b] * 5 + (tid - 54)];
        else               v = emb3[cid3[b] * 2 + (tid - 59)];
        feat[tid] = fast_tanh(v);
    }
    __syncthreads();
    int n = tid;                       // 256 threads = 256 cols
    float acc = b_all[n];              // fold layer-1 bias here
    for (int j = 0; j < 61; ++j) acc += feat[j] * Wih0[(1 + j) * 256 + n];
    catproj[b * 256 + n] = acc;
}

__global__ void timeproj_kernel(const int* __restrict__ tid0, const int* __restrict__ tid1,
                                const int* __restrict__ tid2,
                                const float* __restrict__ temb0, const float* __restrict__ temb1,
                                const float* __restrict__ temb2,
                                const float* __restrict__ Wih0, float* __restrict__ timeproj)
{
    __shared__ float feat[26];
    int t = blockIdx.x;
    int tid = threadIdx.x;
    if (tid < 26) {
        float v;
        if (tid < 4)       v = temb0[tid0[t] * 4 + tid];
        else if (tid < 10) v = temb1[tid1[t] * 6 + (tid - 4)];
        else               v = temb2[tid2[t] * 16 + (tid - 10)];
        feat[tid] = fast_tanh(v);
    }
    __syncthreads();
    int n = tid;
    float acc = 0.f;
    for (int j = 0; j < 26; ++j) acc += feat[j] * Wih0[(62 + j) * 256 + n];
    timeproj[t * 256 + n] = acc;
}

// =======================  exponential-smoothing scan  =======================
__global__ void es_kernel(const float* __restrict__ y, const int* __restrict__ indexes,
                          const float* __restrict__ alpha_raw, const float* __restrict__ gamma_raw,
                          const float* __restrict__ seas_init,
                          float* __restrict__ normv, float* __restrict__ targets)
{
    int b = blockIdx.x * blockDim.x + threadIdx.x;
    if (b >= BB) return;
    int idx = indexes[b];
    float alpha = fast_sig(alpha_raw[idx]);
    float gamma = fast_sig(gamma_raw[idx]);
    float s[7];
    #pragma unroll
    for (int j = 0; j < 7; ++j)
        s[j] = __builtin_amdgcn_exp2f(1.44269504f * seas_init[idx * 7 + j]);
    float y0 = y[b * 512 + 0];
    float prev_level = y0 / s[0];
    float buf[7];
    #pragma unroll
    for (int j = 0; j < 6; ++j) buf[j] = s[j + 1];
    buf[6] = s[0];
    normv[0 * BB + b] = y0 / (s[0] * prev_level);
    for (int t = 1; t < 512; ++t) {
        float yt = y[b * 512 + t];
        float sh = buf[0];
        float nl = alpha * (yt / sh) + (1.f - alpha) * prev_level;
        float sn = gamma * (yt / nl) + (1.f - gamma) * sh;
        #pragma unroll
        for (int j = 0; j < 6; ++j) buf[j] = buf[j + 1];
        buf[6] = sn;
        if (t < T1) normv[t * BB + b] = yt / (sh * nl);
        targets[(t - 1) * BB + b] = yt / (sh * prev_level);
        prev_level = nl;
    }
}

// =======================  weight pre-pack (f32 -> bf16 fragments)  =======================
__global__ void prepack_kernel(const float* __restrict__ Whh_all, const float* __restrict__ Wih_rest,
                               const float* __restrict__ Wlin,
                               bf16_t* pk1, bf16_t* pk2, bf16_t* pk3, bf16_t* pk4, bf16_t* pklin)
{
    int mat = blockIdx.y;
    int idx = blockIdx.x * blockDim.x + threadIdx.x;
    int nel = (mat == 0) ? 16384 : (mat == 4 ? 4096 : 32768);
    if (idx >= nel) return;
    int e    = idx & 15;
    int lane = (idx >> 4) & 31;
    int fj   = idx >> 9;
    int nt   = (mat == 4) ? 4 : NT;
    int jt   = fj % nt;
    int kf   = fj / nt;
    int k    = kf * 32 + kmap2(e >> 1, lane >> 4) + (e & 1);
    int n    = jt * 16 + (lane & 15);
    float val;
    if (mat == 0)      val = Whh_all[k * 256 + n];
    else if (mat == 4) val = Wlin[k * 64 + n];
    else {
        int l = mat;   // layer 2..4 -> 1..3
        val = (k < 64) ? Wih_rest[(size_t)(l - 1) * 64 * 256 + k * 256 + n]
                       : Whh_all [(size_t)l * 64 * 256 + (k - 64) * 256 + n];
    }
    bf16_t* dst = (mat == 0) ? pk1 : (mat == 1) ? pk2 : (mat == 2) ? pk3 : (mat == 3) ? pk4 : pklin;
    dst[idx] = (bf16_t)val;
}

// =======================  layer 1 (d=1), fused rank-1 + table input  =======================
__global__ void __launch_bounds__(128, 1) lstm_layer1_kernel(
    const float* __restrict__ normv, const float* __restrict__ catproj,
    const float* __restrict__ timeproj, const float* __restrict__ W0row,
    const bf16_t* __restrict__ wpk, bf16_t* __restrict__ hout)
{
    __shared__ __align__(16) bf16_t hT_all[WPB][16 * LDT];
    int w = threadIdx.x >> 5, lane = threadIdx.x & 31;
    int half = lane >> 4, ln = lane & 15;
    int unit = blockIdx.x * WPB + w;        // 64 units, one batch strip each
    int b0 = unit * 16;
    bf16_t* hT = hT_all[w];
    for (int i = lane; i < 16 * LDT; i += 32) hT[i] = (bf16_t)0.f;

    // ---- resident weights: 32 B-fragments = 256 VGPRs ----
    const v16bf* wp = (const v16bf*)wpk;
    v16bf bw[2 * NT];
    #pragma unroll
    for (int f = 0; f < 2 * NT; ++f) bw[f] = wp[f * 32 + lane];

    float w0r[NT], base[NT][8];
    #pragma unroll
    for (int jt = 0; jt < NT; ++jt) {
        int n = jt * 16 + ln;
        w0r[jt] = W0row[n];
        #pragma unroll
        for (int v = 0; v < 8; ++v)
            base[jt][v] = catproj[(b0 + v + 8 * half) * 256 + n];
    }
    v8f cc[4];
    #pragma unroll
    for (int c = 0; c < 4; ++c)
        #pragma unroll
        for (int i = 0; i < 8; ++i) cc[c][i] = 0.f;

    for (int t = 0; t < T1; ++t) {
        FragU a0, a1;
        #pragma unroll
        for (int i = 0; i < 8; ++i) {
            int K2 = kmap2(i, half);
            a0.u[i] = *(const unsigned*)(hT + ln * LDT + K2);
            a1.u[i] = *(const unsigned*)(hT + ln * LDT + 32 + K2);
        }
        v8f acc[NT];
        #pragma unroll
        for (int jt = 0; jt < NT; ++jt) {
            #pragma unroll
            for (int i = 0; i < 8; ++i) acc[jt][i] = 0.f;
            acc[jt] = __builtin_amdgcn_wmma_f32_16x16x32_bf16(false, a0.v, false, bw[0 * NT + jt],
                                                              (short)0, acc[jt], false, false);
            acc[jt] = __builtin_amdgcn_wmma_f32_16x16x32_bf16(false, a1.v, false, bw[1 * NT + jt],
                                                              (short)0, acc[jt], false, false);
        }
        float tp[NT], nm[8];
        #pragma unroll
        for (int jt = 0; jt < NT; ++jt) tp[jt] = timeproj[t * 256 + jt * 16 + ln];
        #pragma unroll
        for (int v = 0; v < 8; ++v) nm[v] = normv[(size_t)t * BB + b0 + v + 8 * half];
        #pragma unroll
        for (int jt = 0; jt < NT; ++jt)
            #pragma unroll
            for (int v = 0; v < 8; ++v)
                acc[jt][v] += nm[v] * w0r[jt] + base[jt][v] + tp[jt];

        #pragma unroll
        for (int ct = 0; ct < 4; ++ct)
            #pragma unroll
            for (int v = 0; v < 8; ++v) {
                float i_ = acc[ct][v],  f_ = acc[ct + 4][v];
                float g_ = acc[ct + 8][v], o_ = acc[ct + 12][v];
                float cn = fast_sig(f_) * cc[ct][v] + fast_sig(i_) * fast_tanh(g_);
                cc[ct][v] = cn;
                float h = fast_sig(o_) * fast_tanh(cn);
                hT[(v + 8 * half) * LDT + ct * 16 + ln] = (bf16_t)h;
            }
        bf16_t* gh = hout + ((size_t)t * BB + b0) * HH;
        #pragma unroll
        for (int q = 0; q < 4; ++q) {
            int li = q * 32 + lane; int row = li >> 3; int c8 = li & 7;
            *(uint4*)(gh + row * HH + c8 * 8) = *(const uint4*)(hT + row * LDT + c8 * 8);
        }
    }
}

// =======================  dilated layers 2..4 (K=128 fused [x|h])  =======================
template <int RESID>
__global__ void __launch_bounds__(128, 1) lstm_dil_kernel(
    const bf16_t* __restrict__ xin, const bf16_t* __restrict__ wpk,
    const float* __restrict__ bias, bf16_t* __restrict__ hout,
    const bf16_t* __restrict__ res, int d)
{
    __shared__ __align__(16) bf16_t hT_all[WPB][16 * LDT];
    int w = threadIdx.x >> 5, lane = threadIdx.x & 31;
    int half = lane >> 4, ln = lane & 15;
    int unit = blockIdx.x * WPB + w;        // 64*d units: (strip, phase)
    int s = unit / d, p = unit % d;
    int b0 = s * 16;
    bf16_t* hT = hT_all[w];
    for (int i = lane; i < 16 * LDT; i += 32) hT[i] = (bf16_t)0.f;

    // ---- resident weights: 64 B-fragments = 512 VGPRs ----
    const v16bf* wp = (const v16bf*)wpk;
    v16bf bw[4 * NT];
    #pragma unroll
    for (int f = 0; f < 4 * NT; ++f) bw[f] = wp[f * 32 + lane];

    float bs[NT];
    #pragma unroll
    for (int jt = 0; jt < NT; ++jt) bs[jt] = bias[jt * 16 + ln];
    v8f cc[4];
    #pragma unroll
    for (int c = 0; c < 4; ++c)
        #pragma unroll
        for (int i = 0; i < 8; ++i) cc[c][i] = 0.f;

    for (int t = p; t < T1; t += d) {
        const bf16_t* gx = xin + ((size_t)t * BB + b0) * HH;
        if (t + d < T1)
            __builtin_prefetch(xin + ((size_t)(t + d) * BB + b0) * HH + lane * 32, 0, 1);

        FragU a[4];
        #pragma unroll
        for (int i = 0; i < 8; ++i) {
            int K2 = kmap2(i, half);
            a[0].u[i] = *(const unsigned*)(gx + ln * HH + K2);
            a[1].u[i] = *(const unsigned*)(gx + ln * HH + 32 + K2);
            a[2].u[i] = *(const unsigned*)(hT + ln * LDT + K2);
            a[3].u[i] = *(const unsigned*)(hT + ln * LDT + 32 + K2);
        }
        v8f acc[NT];
        #pragma unroll
        for (int jt = 0; jt < NT; ++jt) {
            #pragma unroll
            for (int i = 0; i < 8; ++i) acc[jt][i] = 0.f;
            #pragma unroll
            for (int kf = 0; kf < 4; ++kf)
                acc[jt] = __builtin_amdgcn_wmma_f32_16x16x32_bf16(false, a[kf].v, false, bw[kf * NT + jt],
                                                                  (short)0, acc[jt], false, false);
        }
        #pragma unroll
        for (int ct = 0; ct < 4; ++ct)
            #pragma unroll
            for (int v = 0; v < 8; ++v) {
                float i_ = acc[ct][v] + bs[ct];
                float f_ = acc[ct + 4][v] + bs[ct + 4];
                float g_ = acc[ct + 8][v] + bs[ct + 8];
                float o_ = acc[ct + 12][v] + bs[ct + 12];
                float cn = fast_sig(f_) * cc[ct][v] + fast_sig(i_) * fast_tanh(g_);
                cc[ct][v] = cn;
                float h = fast_sig(o_) * fast_tanh(cn);
                hT[(v + 8 * half) * LDT + ct * 16 + ln] = (bf16_t)h;
            }
        bf16_t* gh = hout + ((size_t)t * BB + b0) * HH;
        #pragma unroll
        for (int q = 0; q < 4; ++q) {
            int li = q * 32 + lane; int row = li >> 3; int c8 = li & 7;
            if (RESID) {
                Pk16 hv, rv, ov;
                hv.q = *(const uint4*)(hT + row * LDT + c8 * 8);
                rv.q = *(const uint4*)(res + ((size_t)t * BB + b0) * HH + row * HH + c8 * 8);
                #pragma unroll
                for (int k2 = 0; k2 < 8; ++k2)
                    ov.b[k2] = (bf16_t)((float)hv.b[k2] + (float)rv.b[k2]);
                *(uint4*)(gh + row * HH + c8 * 8) = ov.q;
            } else {
                *(uint4*)(gh + row * HH + c8 * 8) = *(const uint4*)(hT + row * LDT + c8 * 8);
            }
        }
    }
}

// =======================  head: tanh(out2@Wlin + blin) @ Wscore + bscore  =======================
__global__ void __launch_bounds__(256) head_kernel(
    const bf16_t* __restrict__ out2, const bf16_t* __restrict__ pklin,
    const float* __restrict__ blin, const float* __restrict__ Wscore,
    const float* __restrict__ bscore, float* __restrict__ pred)
{
    int w = threadIdx.x >> 5, lane = threadIdx.x & 31;
    int half = lane >> 4, ln = lane & 15;
    int strip = blockIdx.x * 8 + w;          // 32704 strips of 16 rows
    int row0 = strip * 16;
    const bf16_t* gx = out2 + (size_t)row0 * HH;
    FragU a0, a1;
    #pragma unroll
    for (int i = 0; i < 8; ++i) {
        int K2 = kmap2(i, half);
        a0.u[i] = *(const unsigned*)(gx + ln * HH + K2);
        a1.u[i] = *(const unsigned*)(gx + ln * HH + 32 + K2);
    }
    const v16bf* wp = (const v16bf*)pklin;
    v8f acc[4];
    #pragma unroll
    for (int jt = 0; jt < 4; ++jt) {
        #pragma unroll
        for (int i = 0; i < 8; ++i) acc[jt][i] = 0.f;
        acc[jt] = __builtin_amdgcn_wmma_f32_16x16x32_bf16(false, a0.v, false, wp[(0 * 4 + jt) * 32 + lane],
                                                          (short)0, acc[jt], false, false);
        acc[jt] = __builtin_amdgcn_wmma_f32_16x16x32_bf16(false, a1.v, false, wp[(1 * 4 + jt) * 32 + lane],
                                                          (short)0, acc[jt], false, false);
    }
    float bl[4], ws[4];
    #pragma unroll
    for (int jt = 0; jt < 4; ++jt) {
        int n = jt * 16 + ln;
        bl[jt] = blin[n];
        ws[jt] = Wscore[n];
    }
    float bsc = bscore[0];
    #pragma unroll
    for (int v = 0; v < 8; ++v) {
        float psum = 0.f;
        #pragma unroll
        for (int jt = 0; jt < 4; ++jt)
            psum += fast_tanh(acc[jt][v] + bl[jt]) * ws[jt];
        psum += __shfl_xor(psum, 1);
        psum += __shfl_xor(psum, 2);
        psum += __shfl_xor(psum, 4);
        psum += __shfl_xor(psum, 8);
        if (ln == 0) pred[row0 + v + 8 * half] = psum + bsc;
    }
}

// ================================  launcher  ================================
extern "C" void kernel_launch(void* const* d_in, const int* in_sizes, int n_in,
                              void* d_out, int out_size, void* d_ws, size_t ws_size,
                              hipStream_t stream)
{
    const float* y         = (const float*)d_in[0];
    const int*   indexes   = (const int*)d_in[1];
    const int*   cid0      = (const int*)d_in[2];
    const int*   cid1      = (const int*)d_in[3];
    const int*   cid2      = (const int*)d_in[4];
    const int*   cid3      = (const int*)d_in[5];
    const int*   tid0      = (const int*)d_in[6];
    const int*   tid1      = (const int*)d_in[7];
    const int*   tid2      = (const int*)d_in[8];
    const float* alpha_raw = (const float*)d_in[9];
    const float* gamma_raw = (const float*)d_in[10];
    const float* seas_init = (const float*)d_in[11];
    const float* emb0      = (const float*)d_in[12];
    const float* emb1      = (const float*)d_in[13];
    const float* emb2      = (const float*)d_in[14];
    const float* emb3      = (const float*)d_in[15];
    const float* temb0     = (const float*)d_in[16];
    const float* temb1     = (const float*)d_in[17];
    const float* temb2     = (const float*)d_in[18];
    const float* Wih0      = (const float*)d_in[19];
    const float* Wih_rest  = (const float*)d_in[20];
    const float* Whh_all   = (const float*)d_in[21];
    const float* b_all     = (const float*)d_in[22];
    const float* Wlin      = (const float*)d_in[23];
    const float* blin      = (const float*)d_in[24];
    const float* Wscore    = (const float*)d_in[25];
    const float* bscore    = (const float*)d_in[26];

    // workspace layout (~132 MiB total)
    char* ws = (char*)d_ws;
    size_t o = 0;
    bf16_t* Wb0    = (bf16_t*)(ws + o); o += (size_t)T1 * BB * HH * 2;   // h1 -> h2 -> out2
    bf16_t* Wb1    = (bf16_t*)(ws + o); o += (size_t)T1 * BB * HH * 2;   // out1
    float* normv   = (float*)(ws + o);  o += (size_t)T1 * BB * 4;
    float* catproj = (float*)(ws + o);  o += 1024 * 256 * 4;
    float* timeprj = (float*)(ws + o);  o += 511 * 256 * 4;
    bf16_t* pk1    = (bf16_t*)(ws + o); o += 16384 * 2;
    bf16_t* pk2    = (bf16_t*)(ws + o); o += 32768 * 2;
    bf16_t* pk3    = (bf16_t*)(ws + o); o += 32768 * 2;
    bf16_t* pk4    = (bf16_t*)(ws + o); o += 32768 * 2;
    bf16_t* pklin  = (bf16_t*)(ws + o); o += 4096 * 2;

    float* pred    = (float*)d_out;
    float* targets = pred + (size_t)T1 * BB;

    catproj_kernel <<<1024, 256, 0, stream>>>(cid0, cid1, cid2, cid3, emb0, emb1, emb2, emb3,
                                              Wih0, b_all, catproj);
    timeproj_kernel<<<511, 256, 0, stream>>>(tid0, tid1, tid2, temb0, temb1, temb2, Wih0, timeprj);
    es_kernel      <<<4, 256, 0, stream>>>(y, indexes, alpha_raw, gamma_raw, seas_init,
                                           normv, targets);
    prepack_kernel <<<dim3(128, 5), 256, 0, stream>>>(Whh_all, Wih_rest, Wlin,
                                                      pk1, pk2, pk3, pk4, pklin);

    lstm_layer1_kernel<<<16, 128, 0, stream>>>(normv, catproj, timeprj, Wih0, pk1, Wb0);
    lstm_dil_kernel<0><<<112, 128, 0, stream>>>(Wb0, pk2, b_all + 256, Wb1, nullptr, 7);   // out1
    lstm_dil_kernel<0><<<224, 128, 0, stream>>>(Wb1, pk3, b_all + 512, Wb0, nullptr, 14);  // h2
    lstm_dil_kernel<1><<<448, 128, 0, stream>>>(Wb0, pk4, b_all + 768, Wb0, Wb1, 28);      // out2 = h4 + out1 (in place, per-unit read-before-write)
    head_kernel       <<<4088, 256, 0, stream>>>(Wb0, pklin, blin, Wscore, bscore, pred);
}
